// MultiTaskDistillation_14705968021606
// MI455X (gfx1250) — compile-verified, matrix operand order checked by reading, not run on hardware
//
#include <hip/hip_runtime.h>
#include <hip/hip_bf16.h>

// MI455X (gfx1250) fused multi-task window attention.
// f16 WMMA (v_wmma_f32_16x16x32_f16) for all GEMMs, LDS-resident per-window
// pipeline, one workgroup (8 wave32) per (task, window).
// N-dimension register blocking (2 tiles / 2 accumulators) so the WMMA pipe
// sees independent dependency chains and A fragments are loaded once.

typedef _Float16 half_t;
typedef __attribute__((ext_vector_type(8)))  _Float16 v8h;
typedef __attribute__((ext_vector_type(16))) _Float16 v16h;
typedef __attribute__((ext_vector_type(8)))  float    v8f;

#define CCH   288
#define HH    112
#define WW    112
#define HW    (HH * WW)
#define CHW   (CCH * HW)
#define WIN   7
#define NTOK  49
#define NPAD  64
#define HEADS 9
#define DHEAD 32

// dynamic LDS layout (bytes); regions are reused across phases:
//  phase1: rawq[64x288 f32] @0, rawk @73728, lnq[64x288 f16] @147456, lnk @184320
//  phase2: Q f16 @0, K f16 @36864, Vt f16 (288x64) @73728   (raw dead)
//  phase3+: S f32 (9x64x64) @110592..258048                 (lnq/lnk dead)
//  phase5: O f16 @0                                         (Q dead)
#define OFF_RAWQ 0
#define OFF_RAWK 73728
#define OFF_LNQ  147456
#define OFF_LNK  184320
#define OFF_Q    0
#define OFF_K    36864
#define OFF_VT   73728
#define OFF_S    110592
#define OFF_O    0
#define SMEM_BYTES 258048

struct KArgs {
  const float* feat[3];
  const float* ng[3];
  const float* nb[3];
  const float* rpb[3];
  const float* bq[3];
  const float* bk[3];
  const float* bv[3];
  const float* bp[3];
  const half_t* w16;   // 3 tasks x {Wq,Wk,Wv,Wp} x 288x288 f16
  float* out;
};

// Load one 16-half WMMA fragment row-slice: halves 0..7 at p, 8..15 at p+16.
// Caller bakes the lane-dependent +8 K offset (lanes 16..31) into p.
__device__ __forceinline__ v16h load_frag(const half_t* p) {
  v8h lo = *(const v8h*)(p);
  v8h hi = *(const v8h*)(p + 16);
  return __builtin_shufflevector(lo, hi, 0, 1, 2, 3, 4, 5, 6, 7,
                                 8, 9, 10, 11, 12, 13, 14, 15);
}

__device__ __forceinline__ v8f wmma_f16(v16h a, v16h b, v8f c) {
  return __builtin_amdgcn_wmma_f32_16x16x32_f16(false, a, false, b,
                                                (short)0, c, false, false);
}

__launch_bounds__(256, 1)
__global__ void mtl_attn_kernel(KArgs A) {
  extern __shared__ char smem[];
  __shared__ float red[64][4][4];          // per-token partial LN stats
  __shared__ float rpbs[169 * HEADS];      // rel-pos bias table

  const int tid  = threadIdx.x;
  const int lane = tid & 31;
  const int wave = tid >> 5;
  const int kh   = (lane >> 4) << 3;       // +8 (K and M) offset for lanes 16..31
  const int ml   = lane & 15;

  const int wid  = blockIdx.x;
  const int task = wid >> 10;              // / 1024
  const int w    = wid & 1023;
  const int b    = w >> 8;
  const int widx = w & 255;
  const int y0   = (widx >> 4) * WIN;
  const int x0   = (widx & 15) * WIN;

  const float* xq = A.feat[task];
  const float* xa = A.feat[task == 0 ? 1 : 0];
  const float* xb = A.feat[task == 2 ? 1 : 2];

  float*  rawq = (float*)(smem + OFF_RAWQ);
  float*  rawk = (float*)(smem + OFF_RAWK);
  half_t* lnq  = (half_t*)(smem + OFF_LNQ);
  half_t* lnk  = (half_t*)(smem + OFF_LNK);
  half_t* Qb   = (half_t*)(smem + OFF_Q);
  half_t* Kb   = (half_t*)(smem + OFF_K);
  half_t* Vt   = (half_t*)(smem + OFF_VT);
  float*  Sb   = (float*)(smem + OFF_S);
  half_t* Ob   = (half_t*)(smem + OFF_O);

  // stage rel-pos bias table into LDS
  for (int i = tid; i < 169 * HEADS; i += 256) rpbs[i] = A.rpb[task][i];

  // ---- Phase 1a: gather window (xq, xk = other two summed), partial stats
  {
    const int tok = tid & 63;
    const int cg  = tid >> 6;              // 4 chunks of 72 channels
    float sq = 0.f, sqq = 0.f, sk = 0.f, skk = 0.f;
    if (tok < NTOK) {
      const int y = y0 + tok / 7, x = x0 + tok % 7;
      const size_t base = (size_t)b * CHW + (size_t)(cg * 72) * HW +
                          (size_t)y * WW + x;
      for (int i = 0; i < 72; ++i) {
        const size_t gi = base + (size_t)i * HW;
        const float q = xq[gi];
        const float k = xa[gi] + xb[gi];
        rawq[tok * CCH + cg * 72 + i] = q;
        rawk[tok * CCH + cg * 72 + i] = k;
        sq += q; sqq += q * q;
        sk += k; skk += k * k;
      }
    }
    red[tok][cg][0] = sq;  red[tok][cg][1] = sqq;
    red[tok][cg][2] = sk;  red[tok][cg][3] = skk;
  }
  __syncthreads();

  // ---- Phase 1b: LayerNorm -> f16 (pad rows zeroed)
  {
    const int tok = tid & 63;
    const int cg  = tid >> 6;
    const float sq  = red[tok][0][0] + red[tok][1][0] + red[tok][2][0] + red[tok][3][0];
    const float sqq = red[tok][0][1] + red[tok][1][1] + red[tok][2][1] + red[tok][3][1];
    const float sk  = red[tok][0][2] + red[tok][1][2] + red[tok][2][2] + red[tok][3][2];
    const float skk = red[tok][0][3] + red[tok][1][3] + red[tok][2][3] + red[tok][3][3];
    const float r288 = 1.f / 288.f;
    const float mq = sq * r288;
    const float iq = rsqrtf(sqq * r288 - mq * mq + 1e-5f);
    const float mk = sk * r288;
    const float ik = rsqrtf(skk * r288 - mk * mk + 1e-5f);
    const float* g  = A.ng[task];
    const float* bb = A.nb[task];
    for (int i = 0; i < 72; ++i) {
      const int c = cg * 72 + i;
      float qv = 0.f, kv = 0.f;
      if (tok < NTOK) {
        const float gg = g[c], bv = bb[c];
        qv = (rawq[tok * CCH + c] - mq) * iq * gg + bv;
        kv = (rawk[tok * CCH + c] - mk) * ik * gg + bv;
      }
      lnq[tok * CCH + c] = (half_t)qv;
      lnk[tok * CCH + c] = (half_t)kv;
    }
  }
  __syncthreads();

  // ---- Phase 2: QKV GEMMs. 108 jobs, each = 16x32 output (2 N-tiles,
  //      2 independent accumulators sharing the A fragment per k-step).
  const half_t* wbase = A.w16 + (size_t)task * 4 * CCH * CCH;
  {
    const float* bqv = A.bq[task];
    const float* bkv = A.bk[task];
    const float* bvv = A.bv[task];
    const float qscale = 0.17677669529663687f;   // 32^-0.5
    for (int job = wave; job < 108; job += 8) {
      const int mat = job / 36;                  // 0=Q,1=K,2=V
      const int t36 = job % 36;
      const int m0  = (t36 / 9) * 16;
      const int np  = (t36 % 9) * 32;            // N-tile pair: np, np+16
      const half_t* src = (mat == 0) ? lnq : lnk;
      const half_t* wm  = wbase + (size_t)mat * CCH * CCH;
      v8f acc0 = {}, acc1 = {};
      for (int ks = 0; ks < 9; ++ks) {
        const int k0 = ks * 32 + kh;
        v16h af = load_frag(src + (m0 + ml) * CCH + k0);
        v16h b0 = load_frag(wm + (np + ml) * CCH + k0);       // B[k=c][n=o]=W[o][c]
        v16h b1 = load_frag(wm + (np + 16 + ml) * CCH + k0);
        acc0 = wmma_f16(af, b0, acc0);
        acc1 = wmma_f16(af, b1, acc1);
      }
      const int c0 = np + ml;
      const int c1 = np + 16 + ml;
      if (mat == 0) {
        const float bi0 = bqv[c0], bi1 = bqv[c1];
#pragma unroll
        for (int r = 0; r < 8; ++r) {
          const int row = (m0 + r + kh) * CCH;
          Qb[row + c0] = (half_t)((acc0[r] + bi0) * qscale);
          Qb[row + c1] = (half_t)((acc1[r] + bi1) * qscale);
        }
      } else if (mat == 1) {
        const float bi0 = bkv[c0], bi1 = bkv[c1];
#pragma unroll
        for (int r = 0; r < 8; ++r) {
          const int row = (m0 + r + kh) * CCH;
          Kb[row + c0] = (half_t)(acc0[r] + bi0);
          Kb[row + c1] = (half_t)(acc1[r] + bi1);
        }
      } else {
        const float bi0 = bvv[c0], bi1 = bvv[c1];
#pragma unroll
        for (int r = 0; r < 8; ++r) {                         // store V transposed
          const int row = m0 + r + kh;
          Vt[c0 * NPAD + row] = (half_t)(acc0[r] + bi0);
          Vt[c1 * NPAD + row] = (half_t)(acc1[r] + bi1);
        }
      }
    }
  }
  __syncthreads();

  // ---- Phase 3: S = Q K^T per head (d=32 == one WMMA k-step), head per wave.
  //      A fragment shared across the 4 N-tiles of each M row-block.
  for (int h = wave; h < HEADS; h += 8) {
    float* Sh = Sb + h * NPAD * NPAD;
    const int kq = h * DHEAD + kh;
    for (int mt = 0; mt < 4; ++mt) {
      const int m0 = mt * 16;
      v16h af = load_frag(Qb + (m0 + ml) * CCH + kq);
#pragma unroll
      for (int nt = 0; nt < 4; ++nt) {
        const int n0 = nt * 16;
        v16h bf = load_frag(Kb + (n0 + ml) * CCH + kq);   // B[k=d][n=tok]=K[tok][d]
        v8f acc = {};
        acc = wmma_f16(af, bf, acc);
#pragma unroll
        for (int r = 0; r < 8; ++r)
          Sh[(m0 + r + kh) * NPAD + n0 + ml] = acc[r];
      }
    }
  }
  __syncthreads();

  // ---- Phase 4: softmax(S + rel-pos bias) -> P (f16, padded cols = 0)
  for (int h = wave; h < HEADS; h += 8) {
    float*  Sh = Sb + h * NPAD * NPAD;
    half_t* Ph = (half_t*)Sh;                // f16 rows at f32 row starts (stride 128)
    for (int rr = 0; rr < 2; ++rr) {
      const int r  = lane + rr * 32;
      const int ry = r / 7, rx = r % 7;
      float mx = -3.0e38f;
      for (int j = 0; j < NTOK; ++j) {
        float bias = 0.f;
        if (r < NTOK) {
          const int idx = (ry - j / 7 + 6) * 13 + (rx - j % 7 + 6);
          bias = rpbs[idx * HEADS + h];
        }
        mx = fmaxf(mx, Sh[r * NPAD + j] + bias);
      }
      float sum = 0.f;
      for (int j = 0; j < NTOK; ++j) {
        float bias = 0.f;
        if (r < NTOK) {
          const int idx = (ry - j / 7 + 6) * 13 + (rx - j % 7 + 6);
          bias = rpbs[idx * HEADS + h];
        }
        const float e = __expf(Sh[r * NPAD + j] + bias - mx);
        Sh[r * NPAD + j] = e;
        sum += e;
      }
      const float inv = 1.f / sum;
      for (int j = 0; j < NTOK; ++j)
        Ph[r * 128 + j] = (half_t)(Sh[r * NPAD + j] * inv);
      for (int j = NTOK; j < NPAD; ++j)
        Ph[r * 128 + j] = (half_t)0.f;
    }
  }
  __syncthreads();

  // ---- Phase 5: O = P V per head (K=64 -> 2 k-steps), concat heads into Ob.
  //      Both k-step A fragments shared across the 2 N-tiles.
  for (int h = wave; h < HEADS; h += 8) {
    const half_t* Ph = (const half_t*)(Sb + h * NPAD * NPAD);
    for (int mt = 0; mt < 4; ++mt) {
      const int m0 = mt * 16;
      v16h a0 = load_frag(Ph + (m0 + ml) * 128 + kh);
      v16h a1 = load_frag(Ph + (m0 + ml) * 128 + 32 + kh);
#pragma unroll
      for (int nt = 0; nt < 2; ++nt) {
        const int colc = h * DHEAD + nt * 16 + ml;
        v16h b0 = load_frag(Vt + colc * NPAD + kh);       // B[k=tok][n=d]=Vt[d][tok]
        v16h b1 = load_frag(Vt + colc * NPAD + 32 + kh);
        v8f acc = {};
        acc = wmma_f16(a0, b0, acc);
        acc = wmma_f16(a1, b1, acc);
#pragma unroll
        for (int r = 0; r < 8; ++r)
          Ob[(m0 + r + kh) * CCH + colc] = (half_t)acc[r];
      }
    }
  }
  __syncthreads();

  // ---- Phase 6: projection + bias + residual -> NCHW fp32 output.
  //      36 jobs, each = 16x32 output (2 N-tiles, 2 accumulators).
  {
    const half_t* wp  = wbase + (size_t)3 * CCH * CCH;
    const float* bpv  = A.bp[task];
    float* outp = A.out + (size_t)task * 4 * CHW;
    for (int job = wave; job < 36; job += 8) {
      const int m0 = (job / 9) * 16;
      const int np = (job % 9) * 32;
      v8f acc0 = {}, acc1 = {};
      for (int ks = 0; ks < 9; ++ks) {
        const int k0 = ks * 32 + kh;
        v16h af = load_frag(Ob + (m0 + ml) * CCH + k0);
        v16h b0 = load_frag(wp + (np + ml) * CCH + k0);
        v16h b1 = load_frag(wp + (np + 16 + ml) * CCH + k0);
        acc0 = wmma_f16(af, b0, acc0);
        acc1 = wmma_f16(af, b1, acc1);
      }
      const int c0 = np + ml;
      const int c1 = np + 16 + ml;
      const float bi0 = bpv[c0], bi1 = bpv[c1];
#pragma unroll
      for (int r = 0; r < 8; ++r) {
        const int row = m0 + r + kh;
        if (row < NTOK) {
          const int y = y0 + row / 7, x = x0 + row % 7;
          const size_t gbase = (size_t)b * CHW + (size_t)y * WW + x;
          const size_t g0 = gbase + (size_t)c0 * HW;
          const size_t g1 = gbase + (size_t)c1 * HW;
          outp[g0] = acc0[r] + bi0 + xq[g0];
          outp[g1] = acc1[r] + bi1 + xq[g1];
        }
      }
    }
  }
}

__global__ void cvt_w_kernel(const float* __restrict__ src,
                             half_t* __restrict__ dst, int n) {
  int i = blockIdx.x * 256 + threadIdx.x;
  if (i < n) dst[i] = (half_t)src[i];
}

extern "C" void kernel_launch(void* const* d_in, const int* in_sizes, int n_in,
                              void* d_out, int out_size, void* d_ws, size_t ws_size,
                              hipStream_t stream) {
  (void)in_sizes; (void)n_in; (void)out_size; (void)ws_size;
  half_t* w16 = (half_t*)d_ws;

  // params per task: [ng, nb, rpb, Wq, bq, Wk, bk, Wv, bv, Wp, bp]
  const int widx[4] = {3, 5, 7, 9};
  for (int t = 0; t < 3; ++t)
    for (int m = 0; m < 4; ++m) {
      const float* src = (const float*)d_in[3 + t * 11 + widx[m]];
      half_t* dst = w16 + (size_t)(t * 4 + m) * CCH * CCH;
      cvt_w_kernel<<<(CCH * CCH + 255) / 256, 256, 0, stream>>>(src, dst, CCH * CCH);
    }

  KArgs A;
  for (int t = 0; t < 3; ++t) {
    const int pb = 3 + t * 11;
    A.feat[t] = (const float*)d_in[t];
    A.ng[t]   = (const float*)d_in[pb + 0];
    A.nb[t]   = (const float*)d_in[pb + 1];
    A.rpb[t]  = (const float*)d_in[pb + 2];
    A.bq[t]   = (const float*)d_in[pb + 4];
    A.bk[t]   = (const float*)d_in[pb + 6];
    A.bv[t]   = (const float*)d_in[pb + 8];
    A.bp[t]   = (const float*)d_in[pb + 10];
  }
  A.w16 = w16;
  A.out = (float*)d_out;

  hipFuncSetAttribute((const void*)mtl_attn_kernel,
                      hipFuncAttributeMaxDynamicSharedMemorySize, SMEM_BYTES);
  // 3 tasks x 1024 windows, 256 threads (8 wave32) each
  mtl_attn_kernel<<<3072, 256, SMEM_BYTES, stream>>>(A);
}
